// ExponentialMovingAverage_2525440770978
// MI455X (gfx1250) — matrix operand order repaired and checked by reading, not compiled, not run on hardware
//
#include <hip/hip_runtime.h>
#include <stdint.h>

// ---------------------------------------------------------------------------
// EMA scan  y[b,t,c] = w_c*x[b,t,c] + (1-w_c)*y[b,t-1,c],  y[b,-1,c]=x[b,0,c]
// B=64, T=8000, C=128, fp32.  Memory-bound: 524MB floor -> 22.5us @ 23.3TB/s.
// 3-pass chunked affine scan (chunk=64 steps), local scans done with
// V_WMMA_F32_16X16X32_F16 prefix-sum matmuls (hi/lo f16 split => fp32-class
// accuracy), chunk tiles staged to LDS by the Tensor Data Mover (TDM).
// ---------------------------------------------------------------------------

typedef __attribute__((ext_vector_type(16))) _Float16    v16h;
typedef __attribute__((ext_vector_type(8)))  float       v8f;
typedef __attribute__((ext_vector_type(4)))  unsigned    v4u;
typedef __attribute__((ext_vector_type(8)))  int         v8i;
typedef __attribute__((ext_vector_type(4)))  int         v4i;

constexpr int Bdim   = 64;
constexpr int Tdim   = 8000;
constexpr int Cdim   = 128;
constexpr int CHUNK  = 64;               // timesteps per block
constexpr int NCHUNK = Tdim / CHUNK;     // 125 (exact)
constexpr int TILES  = CHUNK / 16;       // 4 WMMA tiles per chunk

__device__ __forceinline__ float lane_bcast(float v, int srcLane) {
  // ds_bpermute_b32: gather from lane srcLane (byte index)
  return __int_as_float(
      __builtin_amdgcn_ds_bpermute(srcLane << 2, __float_as_int(v)));
}

// Issue a TDM async load of a (rows x 128) f32 tile (row stride 128 elems)
// into LDS offset 0.  Descriptor per CDNA5 ISA §8.3/§8.4 (2D: groups 2,3 zero).
// This toolchain exposes the 6-arg builtin: the extra int32x8 SGPR-group is
// unused for a plain 2D load and is passed zero-filled.
__device__ __forceinline__ void tdm_load_tile(const float* gsrc, int rows) {
  uint64_t ga = (uint64_t)(uintptr_t)gsrc;
  v4u g0;
  g0[0] = 1u;                                   // count=1, user descriptor
  g0[1] = 0u;                                   // lds_addr = 0 (tile is the only LDS var)
  g0[2] = (unsigned)ga;                         // global_addr[31:0]
  g0[3] = (unsigned)((ga >> 32) & 0x01FFFFFFu)  // global_addr[56:32]
        | (2u << 30);                           // type = 2 ("image")
  v8i g1;
  g1[0] = (int)(2u << 16);                      // data_size=2 -> 4 bytes/elem
  g1[1] = (int)(128u << 16);                    // tensor_dim0 = 128 (lo16)
  g1[2] = (int)(((unsigned)rows & 0xFFFFu) << 16); // dim0 hi=0 | tensor_dim1 lo16
  g1[3] = (int)(((unsigned)rows >> 16) | (128u << 16)); // dim1 hi | tile_dim0=128
  g1[4] = (int)((unsigned)rows & 0xFFFFu);      // tile_dim1 = rows, tile_dim2 = 0
  g1[5] = 128;                                  // tensor_dim0_stride (lo32) = 128
  g1[6] = 0;
  g1[7] = 0;
  v4i g2 = {0, 0, 0, 0};                        // 2D tensor: groups 2/3 unused
  v4i g3 = {0, 0, 0, 0};
  v8i g4 = {0, 0, 0, 0, 0, 0, 0, 0};            // extra SGPR group (unused)
  __builtin_amdgcn_tensor_load_to_lds(g0, g1, g2, g3, g4, 0);
}

// ---------------------------------------------------------------------------
// Pass 1 / Pass 3 kernel.  Block = (chunk k, batch b), 256 threads = 8 waves,
// wave wv owns channels [16*wv, 16*wv+16).  Each wave walks its chunk in
// 16-step WMMA tiles:
//   S = L1 @ Z   (prefix sum; Z[j][c] = x[t0+j][c] * q_c^-j, hi/lo f16 split)
//   y_i = q^i * (w*S_i + q*a);   carry a <- y_15 (ds_bpermute broadcast)
// WRITE_OUT=false: only emit chunk summary r (local y_63 with a=0).
// WRITE_OUT=true : read true carry from ws, write y to out.
// ---------------------------------------------------------------------------
template <bool WRITE_OUT>
__global__ void __launch_bounds__(256)
ema_chunk_kernel(const float* __restrict__ x, const float* __restrict__ wts,
                 float* __restrict__ out, float* __restrict__ r_ws,
                 const float* __restrict__ carry_ws) {
  __shared__ float tile[CHUNK * Cdim];   // 32 KB, at LDS offset 0

  const int k    = blockIdx.x;
  const int b    = blockIdx.y;
  const int wv   = threadIdx.x >> 5;
  const int lane = threadIdx.x & 31;
  const int n    = lane & 15;            // channel-in-group == WMMA M/N index
  const int h    = lane >> 4;            // lane half
  const int c    = wv * 16 + n;

  // One wave issues the async tensor DMA for the whole block tile.
  if (threadIdx.x < 32) {
    const float* src = x + ((size_t)b * Tdim + (size_t)k * CHUNK) * Cdim;
    tdm_load_tile(src, CHUNK);
    __builtin_amdgcn_s_wait_tensorcnt(0);
  }
  __syncthreads();

  // Per-channel EMA parameters.
  float w = wts[c];
  w = fminf(fmaxf(w, 0.0f), 1.0f);
  const float q  = 1.0f - w;
  const float qi = (q > 0.0f) ? (1.0f / q) : 0.0f;  // (w==1 degenerates; N/A here)

  float qinvPow[16];                     // q^-e, e=0..15 (amp <= q^-15, ~1.84 @ w=0.04)
  qinvPow[0] = 1.0f;
#pragma unroll
  for (int e = 1; e < 16; ++e) qinvPow[e] = qinvPow[e - 1] * qi;
  float qPow[8];                         // q^v, v=0..7
  qPow[0] = 1.0f;
#pragma unroll
  for (int v = 1; v < 8; ++v) qPow[v] = qPow[v - 1] * q;
  const float qh = h ? (qPow[7] * q) : 1.0f;   // q^(8h)

  // A = [L1 | L1] (16x32 f16): A[i][kk] = ((kk&15) <= i).  16-bit A layout:
  // lane half h: elem e -> K=16h+e (contiguous, matching B).
  v16h amat;
#pragma unroll
  for (int e = 0; e < 16; ++e) {
    int kk  = 16 * h + e;
    amat[e] = ((kk & 15) <= n) ? (_Float16)1.0f : (_Float16)0.0f;
  }

  // Carry-in: pass 3 reads the true per-chunk carry; pass 1 uses 0.
  float a = WRITE_OUT ? carry_ws[((size_t)b * NCHUNK + k) * Cdim + c] : 0.0f;

#pragma unroll
  for (int tt = 0; tt < TILES; ++tt) {
    // B (32x16 f16): lanes h=0 supply rows K=0..15 = f16_hi(z),
    //                lanes h=1 supply rows K=16..31 = f16_lo(z).
    v16h bmat;
#pragma unroll
    for (int e = 0; e < 16; ++e) {
      float    z  = tile[(tt * 16 + e) * Cdim + c] * qinvPow[e];
      _Float16 hi = (_Float16)z;
      bmat[e]     = h ? (_Float16)(z - (float)hi) : hi;
    }
    v8f s = {};
    s = __builtin_amdgcn_wmma_f32_16x16x32_f16(false, amat, false, bmat,
                                               (short)0, s, false, false);
    // y_i = q^i * (w*S_i + q*a), i = v + 8h  (D layout: VGPR v, lane half h)
    const float qa = q * a;
    float y[8];
#pragma unroll
    for (int v = 0; v < 8; ++v) y[v] = (qh * qPow[v]) * fmaf(w, s[v], qa);

    if (WRITE_OUT) {
      const size_t tbase = (size_t)b * Tdim + (size_t)k * CHUNK + tt * 16;
#pragma unroll
      for (int v = 0; v < 8; ++v) out[(tbase + v + 8 * h) * Cdim + c] = y[v];
    }
    // Next carry = y[15][c], held in lane 16+n at v=7; broadcast to both halves.
    a = lane_bcast(y[7], 16 + n);
  }

  if (!WRITE_OUT && h == 0)
    r_ws[((size_t)b * NCHUNK + k) * Cdim + c] = a;   // chunk summary r_k
}

// ---------------------------------------------------------------------------
// Pass 2: sequential scan over the 125 chunk summaries per (b,c):
//   carry[0] = x[b,0,c];  carry[k] = q^64 * carry[k-1] + r[k-1]
// ---------------------------------------------------------------------------
__global__ void __launch_bounds__(128)
ema_carry_scan(const float* __restrict__ x, const float* __restrict__ wts,
               const float* __restrict__ r_ws, float* __restrict__ carry_ws) {
  const int b = blockIdx.x;
  const int c = threadIdx.x;
  float w = wts[c];
  w = fminf(fmaxf(w, 0.0f), 1.0f);
  const float q   = 1.0f - w;
  const float q2  = q * q,   q4  = q2 * q2,  q8 = q4 * q4;
  const float q16 = q8 * q8, q32 = q16 * q16;
  const float qL  = q32 * q32;                     // q^64
  float a = x[(size_t)b * Tdim * Cdim + c];        // init = x[b,0,:]
  carry_ws[((size_t)b * NCHUNK + 0) * Cdim + c] = a;
  for (int k = 1; k < NCHUNK; ++k) {
    a = fmaf(qL, a, r_ws[((size_t)b * NCHUNK + (k - 1)) * Cdim + c]);
    carry_ws[((size_t)b * NCHUNK + k) * Cdim + c] = a;
  }
}

// ---------------------------------------------------------------------------
// Fallback (used only if d_ws is too small): one lane per (b,c) chain.
// ---------------------------------------------------------------------------
__global__ void ema_fallback(const float* __restrict__ x,
                             const float* __restrict__ wts,
                             float* __restrict__ out) {
  const int idx = blockIdx.x * blockDim.x + threadIdx.x;
  if (idx >= Bdim * Cdim) return;
  const int b = idx / Cdim, c = idx % Cdim;
  float w = wts[c];
  w = fminf(fmaxf(w, 0.0f), 1.0f);
  const float q = 1.0f - w;
  const float* xp = x + (size_t)b * Tdim * Cdim + c;
  float* yp = out + (size_t)b * Tdim * Cdim + c;
  float acc = xp[0];
  for (int t = 0; t < Tdim; ++t) {
    acc = fmaf(w, xp[(size_t)t * Cdim], q * acc);
    yp[(size_t)t * Cdim] = acc;
  }
}

extern "C" void kernel_launch(void* const* d_in, const int* in_sizes, int n_in,
                              void* d_out, int out_size, void* d_ws,
                              size_t ws_size, hipStream_t stream) {
  const float* x   = (const float*)d_in[0];
  const float* wts = (const float*)d_in[1];
  float*       out = (float*)d_out;

  const size_t summaries = (size_t)Bdim * NCHUNK * Cdim;     // 1.024M floats
  const size_t need      = 2 * summaries * sizeof(float);    // ~8.2 MB

  if (ws_size < need) {  // defensive: scratch too small -> correct slow path
    ema_fallback<<<(Bdim * Cdim + 255) / 256, 256, 0, stream>>>(x, wts, out);
    return;
  }

  float* r_ws     = (float*)d_ws;
  float* carry_ws = r_ws + summaries;

  dim3 grid(NCHUNK, Bdim), blk(256);
  ema_chunk_kernel<false><<<grid, blk, 0, stream>>>(x, wts, out, r_ws, carry_ws);
  ema_carry_scan<<<dim3(Bdim), dim3(Cdim), 0, stream>>>(x, wts, r_ws, carry_ws);
  ema_chunk_kernel<true><<<grid, blk, 0, stream>>>(x, wts, out, r_ws, carry_ws);
}
